// matcher_78262894068291
// MI455X (gfx1250) — compile-verified, matrix operand order checked by reading, not run on hardware
//
#include <hip/hip_runtime.h>
#include <hip/hip_fp16.h>
#include <math.h>

// ---------------------------------------------------------------------------
// SuperGlue-style matcher forward for MI455X (gfx1250, wave32).
// All matmuls use v_wmma_f32_16x16x32_f16 (f32 accumulate); everything else
// (inorm, bn, softmax_eps, Sinkhorn logsumexp) is fp32 VALU.
// K-loops guard-free (K % 32 == 0; K==2 layers use a scalar kernel).
// Sinkhorn: coupling matrix padded to 16B-aligned rows + one-time transpose
// so all 200 logsumexp sweeps read contiguous float4 rows.
// ---------------------------------------------------------------------------

typedef __attribute__((ext_vector_type(16))) _Float16 v16h;
typedef __attribute__((ext_vector_type(8)))  float    v8f;

#define DEVINL __device__ __forceinline__

static constexpr int WROW = 1025;   // sinkhorn coupling width (N+1)
static constexpr int CPAD = 1040;   // padded row stride (4160 B, 16B aligned)

// CDNA5 16-bit A/B 16x32 fragment: lane<16 holds K kbase+0..7 (elems 0..7)
// and kbase+16..23 (elems 8..15), kbase = 0; lane>=16: kbase = 8.
DEVINL int kmap_lane(int lane, int e) {
  return ((e < 8) ? e : (e + 8)) + ((lane < 16) ? 0 : 8);
}

DEVINL v8f wmma_f16(v16h a, v16h b, v8f c) {
  return __builtin_amdgcn_wmma_f32_16x16x32_f16(false, a, false, b,
                                                (short)0, c, false, false);
}

// pack two contiguous 8-float runs (k=kbase+0..7, kbase+16..23) into a fragment
DEVINL v16h pack16(float4 a0, float4 a1, float4 a2, float4 a3) {
  v16h d;
  d[0]  = (_Float16)a0.x; d[1]  = (_Float16)a0.y; d[2]  = (_Float16)a0.z; d[3]  = (_Float16)a0.w;
  d[4]  = (_Float16)a1.x; d[5]  = (_Float16)a1.y; d[6]  = (_Float16)a1.z; d[7]  = (_Float16)a1.w;
  d[8]  = (_Float16)a2.x; d[9]  = (_Float16)a2.y; d[10] = (_Float16)a2.z; d[11] = (_Float16)a2.w;
  d[12] = (_Float16)a3.x; d[13] = (_Float16)a3.y; d[14] = (_Float16)a3.z; d[15] = (_Float16)a3.w;
  return d;
}

// ---------------------------------------------------------------------------
// Generic conv GEMM:  Y[M x L] = epilogue( sum_k W[m,k] * X[k,l] )
// Requires K % 32 == 0.  A (weights) loaded with float4; B batched scalar.
// ---------------------------------------------------------------------------
template <bool TSCALE, bool TSHIFT, bool TRELU, bool TBASE>
__global__ void k_conv_gemm(const float* __restrict__ W, const float* __restrict__ X,
                            float* __restrict__ Y,
                            const float* __restrict__ scale,
                            const float* __restrict__ shift,
                            const float* __restrict__ base,
                            int M, int K, int L)
{
  int lane = threadIdx.x & 31;
  int wave = threadIdx.x >> 5;
  int tile = blockIdx.x * (blockDim.x >> 5) + wave;
  int ntl  = L >> 4;
  int mt = tile / ntl;
  int nt = tile % ntl;
  if (mt * 16 >= M) return;                  // wave-uniform exit
  int arow = mt * 16 + (lane & 15);
  int bcol = nt * 16 + (lane & 15);
  int kbase = (lane < 16) ? 0 : 8;
  v8f acc = {};
  for (int k0 = 0; k0 < K; k0 += 32) {
    const float* wp = W + (size_t)arow * K + k0 + kbase;
    float4 a0 = ((const float4*)wp)[0];
    float4 a1 = ((const float4*)wp)[1];
    float4 a2 = ((const float4*)(wp + 16))[0];
    float4 a3 = ((const float4*)(wp + 16))[1];
    float bf[16];
#pragma unroll
    for (int e = 0; e < 16; ++e)
      bf[e] = X[(size_t)(k0 + kmap_lane(lane, e)) * L + bcol];
    v16h a = pack16(a0, a1, a2, a3);
    v16h b;
#pragma unroll
    for (int e = 0; e < 16; ++e) b[e] = (_Float16)bf[e];
    acc = wmma_f16(a, b, acc);
  }
  // epilogue: 8 contiguous output rows per lane -> float4 coefficient loads
  int roff = mt * 16 + ((lane < 16) ? 0 : 8);
  float scv[8], shv[8], bsv[8];
  if (TSCALE) {
    float4 s0 = ((const float4*)(scale + roff))[0];
    float4 s1 = ((const float4*)(scale + roff))[1];
    scv[0] = s0.x; scv[1] = s0.y; scv[2] = s0.z; scv[3] = s0.w;
    scv[4] = s1.x; scv[5] = s1.y; scv[6] = s1.z; scv[7] = s1.w;
  }
  if (TSHIFT) {
    float4 s0 = ((const float4*)(shift + roff))[0];
    float4 s1 = ((const float4*)(shift + roff))[1];
    shv[0] = s0.x; shv[1] = s0.y; shv[2] = s0.z; shv[3] = s0.w;
    shv[4] = s1.x; shv[5] = s1.y; shv[6] = s1.z; shv[7] = s1.w;
  }
  if (TBASE) {
#pragma unroll
    for (int r = 0; r < 8; ++r) bsv[r] = base[(size_t)(roff + r) * L + bcol];
  }
#pragma unroll
  for (int r = 0; r < 8; ++r) {
    float v = acc[r];
    if (TSCALE) v *= scv[r];
    if (TSHIFT) v += shv[r];
    if (TRELU)  v = fmaxf(v, 0.0f);
    if (TBASE)  v += bsv[r];
    Y[(size_t)(roff + r) * L + bcol] = v;
  }
}

// 2-input-channel conv + bn + relu (first encoder layers; WMMA pointless at K=2)
__global__ void k_conv2(const float* __restrict__ W, const float* __restrict__ X,
                        float* __restrict__ Y,
                        const float* __restrict__ scale, const float* __restrict__ shift,
                        int Mo, int L)
{
  int i = blockIdx.x * blockDim.x + threadIdx.x;
  if (i >= Mo * L) return;
  int m = i / L, l = i % L;
  float v = W[m * 2 + 0] * X[l] + W[m * 2 + 1] * X[L + l];
  v = v * scale[m] + shift[m];
  Y[i] = fmaxf(v, 0.0f);
}

// ---------------------------------------------------------------------------
// XtY outer product:  O[n,m] = scale * sum_c X[c,n]*Y[c,m]  (- eye)
// Kc % 32 == 0 (always 128 here).  X,Y row strides ldx/ldy; O stride ldo.
// ---------------------------------------------------------------------------
__global__ void k_xty(const float* __restrict__ X, const float* __restrict__ Y,
                      float* __restrict__ O, int Kc, int ldx, int ldy, int ldo,
                      float scale, int minus_eye)
{
  const int L = 1024;
  int lane = threadIdx.x & 31;
  int wave = threadIdx.x >> 5;
  int tile = blockIdx.x * (blockDim.x >> 5) + wave;
  const int ntl = L / 16;                    // 64
  int nt = tile / ntl;
  int mt = tile % ntl;
  int nrow = nt * 16 + (lane & 15);
  int mcol = mt * 16 + (lane & 15);
  v8f acc = {};
  for (int k0 = 0; k0 < Kc; k0 += 32) {
    float af[16], bf[16];
#pragma unroll
    for (int e = 0; e < 16; ++e) {
      int k = k0 + kmap_lane(lane, e);
      af[e] = X[(size_t)k * ldx + nrow];
      bf[e] = Y[(size_t)k * ldy + mcol];
    }
    v16h a, b;
#pragma unroll
    for (int e = 0; e < 16; ++e) { a[e] = (_Float16)af[e]; b[e] = (_Float16)bf[e]; }
    acc = wmma_f16(a, b, acc);
  }
#pragma unroll
  for (int r = 0; r < 8; ++r) {
    int n = nt * 16 + r + ((lane < 16) ? 0 : 8);
    float v = acc[r] * scale;
    if (minus_eye && n == mcol) v -= 1.0f;
    O[(size_t)n * ldo + mcol] = v;
  }
}

// ---------------------------------------------------------------------------
// Attention scores per head:
//   S[h,n,m] = sc * sum_d Q[(d*4+h),n] * Kt[(d*4+h),m]
//            + lrelu(lam[h]*R[n,m] + rbias[h])      (RT: R[m,n])
// ---------------------------------------------------------------------------
template <bool RT>
__global__ void k_attn_scores(const float* __restrict__ Q, const float* __restrict__ Kt,
                              const float* __restrict__ R,
                              const float* __restrict__ lam,
                              const float* __restrict__ rbias,
                              float* __restrict__ S, float sc)
{
  const int L = 1024;
  int lane = threadIdx.x & 31;
  int wave = threadIdx.x >> 5;
  int h = blockIdx.z;
  int tile = blockIdx.x * (blockDim.x >> 5) + wave;
  const int ntl = L / 16;
  int nt = tile / ntl;
  int mt = tile % ntl;
  int nrow = nt * 16 + (lane & 15);
  int mcol = mt * 16 + (lane & 15);
  float af[16], bf[16];
#pragma unroll
  for (int e = 0; e < 16; ++e) {
    int d = kmap_lane(lane, e);
    af[e] = Q[(size_t)(d * 4 + h) * L + nrow];
    bf[e] = Kt[(size_t)(d * 4 + h) * L + mcol];
  }
  v16h a, b;
#pragma unroll
  for (int e = 0; e < 16; ++e) { a[e] = (_Float16)af[e]; b[e] = (_Float16)bf[e]; }
  v8f acc = {};
  acc = wmma_f16(a, b, acc);
  int roff = nt * 16 + ((lane < 16) ? 0 : 8);
  float rvv[8];
  if (RT) {
    const float* rp = R + (size_t)mcol * L + roff;   // contiguous -> float4
    float4 r0 = ((const float4*)rp)[0];
    float4 r1 = ((const float4*)rp)[1];
    rvv[0] = r0.x; rvv[1] = r0.y; rvv[2] = r0.z; rvv[3] = r0.w;
    rvv[4] = r1.x; rvv[5] = r1.y; rvv[6] = r1.z; rvv[7] = r1.w;
  } else {
#pragma unroll
    for (int r = 0; r < 8; ++r) rvv[r] = R[(size_t)(roff + r) * L + mcol];
  }
  float la = lam[h], rb = rbias[h];
#pragma unroll
  for (int r = 0; r < 8; ++r) {
    float t = la * rvv[r] + rb;
    t = (t >= 0.0f) ? t : 0.1f * t;
    S[((size_t)h * L + roff + r) * L + mcol] = acc[r] * sc + t;
  }
}

// ---------------------------------------------------------------------------
// AV per head:  AV[(d*4+h),n] = sum_m P[h,n,m] * V[(d*4+h),m]
// K = 1024; both operands contiguous in m -> fully float4-vectorized loads.
// ---------------------------------------------------------------------------
__global__ void k_attn_av(const float* __restrict__ P, const float* __restrict__ V,
                          float* __restrict__ AV)
{
  const int L = 1024;
  int lane = threadIdx.x & 31;
  int wave = threadIdx.x >> 5;
  int h = blockIdx.z;
  int tile = blockIdx.x * (blockDim.x >> 5) + wave;
  const int ntl = L / 16;                    // col tiles; 2 row tiles (d = 0..31)
  int dt = tile / ntl;
  int nt = tile % ntl;
  int drow = dt * 16 + (lane & 15);
  int ncol = nt * 16 + (lane & 15);
  int kbase = (lane < 16) ? 0 : 8;
  const float* vrow = V + (size_t)(drow * 4 + h) * L;
  const float* prow = P + ((size_t)h * L + ncol) * L;
  v8f acc = {};
  for (int m0 = 0; m0 < L; m0 += 32) {
    const float* vp = vrow + m0 + kbase;
    const float* pp = prow + m0 + kbase;
    float4 a0 = ((const float4*)vp)[0];
    float4 a1 = ((const float4*)vp)[1];
    float4 a2 = ((const float4*)(vp + 16))[0];
    float4 a3 = ((const float4*)(vp + 16))[1];
    float4 b0 = ((const float4*)pp)[0];
    float4 b1 = ((const float4*)pp)[1];
    float4 b2 = ((const float4*)(pp + 16))[0];
    float4 b3 = ((const float4*)(pp + 16))[1];
    acc = wmma_f16(pack16(a0, a1, a2, a3), pack16(b0, b1, b2, b3), acc);
  }
#pragma unroll
  for (int r = 0; r < 8; ++r) {
    int d = dt * 16 + r + ((lane < 16) ? 0 : 8);
    AV[(size_t)(d * 4 + h) * L + ncol] = acc[r];
  }
}

// ------------------------- elementwise / reduction kernels -----------------

__global__ void k_normalize_t(const float* __restrict__ desc, float* __restrict__ D,
                              int Cc, int ldd, int col0)
{
  int n = blockIdx.x;
  int t = threadIdx.x;                       // blockDim == Cc == 128
  __shared__ float red[128];
  float v = desc[(size_t)n * Cc + t];
  red[t] = v * v;
  __syncthreads();
  for (int s = 64; s > 0; s >>= 1) { if (t < s) red[t] += red[t + s]; __syncthreads(); }
  float inv = 1.0f / fmaxf(sqrtf(red[0]), 1e-12f);
  D[(size_t)t * ldd + col0 + n] = v * inv;
}

__global__ void k_build_ex(const float* __restrict__ x1, const float* __restrict__ x2,
                           float* __restrict__ ex)
{
  int n = blockIdx.x * blockDim.x + threadIdx.x;
  if (n >= 1024) return;
  ex[n]               = x1[n * 2 + 0];
  ex[2048 + n]        = x1[n * 2 + 1];
  ex[1024 + n]        = x2[n * 2 + 0];
  ex[2048 + 1024 + n] = x2[n * 2 + 1];
}

__global__ void k_fuse_bn(const float* __restrict__ g, const float* __restrict__ b,
                          const float* __restrict__ bias,
                          float* __restrict__ scale, float* __restrict__ shift, int n)
{
  int i = blockIdx.x * blockDim.x + threadIdx.x;
  if (i >= n) return;
  float gs = g[i] * rsqrtf(1.0f + 1e-5f);
  scale[i] = gs;
  shift[i] = (bias ? bias[i] * gs : 0.0f) + b[i];
}

// per-channel instance norm over L=2048, then bn(g,b), then relu.
// 256 threads x 2 float4 held in registers across the stats reduction.
__global__ void k_inorm_bn_relu(const float* __restrict__ X, float* __restrict__ Y,
                                const float* __restrict__ g, const float* __restrict__ b)
{
  const int L = 2048;
  int c = blockIdx.x;
  int t = threadIdx.x;                       // 256
  __shared__ float s1[256], s2[256];
  const float4* xp = (const float4*)(X + (size_t)c * L);
  float4 x0 = xp[t];
  float4 x1 = xp[t + 256];
  float sum  = x0.x + x0.y + x0.z + x0.w + x1.x + x1.y + x1.z + x1.w;
  float sum2 = x0.x*x0.x + x0.y*x0.y + x0.z*x0.z + x0.w*x0.w
             + x1.x*x1.x + x1.y*x1.y + x1.z*x1.z + x1.w*x1.w;
  s1[t] = sum; s2[t] = sum2;
  __syncthreads();
  for (int s = 128; s > 0; s >>= 1) {
    if (t < s) { s1[t] += s1[t + s]; s2[t] += s2[t + s]; }
    __syncthreads();
  }
  float mu = s1[0] / L;
  float var = s2[0] / L - mu * mu;
  float rsv = rsqrtf(var + 1e-5f);
  float gs = g[c] * rsqrtf(1.0f + 1e-5f);
  float bb = b[c];
  float a = rsv * gs;
  float d = bb - mu * a;
  float4 y0, y1;
  y0.x = fmaxf(x0.x * a + d, 0.0f); y0.y = fmaxf(x0.y * a + d, 0.0f);
  y0.z = fmaxf(x0.z * a + d, 0.0f); y0.w = fmaxf(x0.w * a + d, 0.0f);
  y1.x = fmaxf(x1.x * a + d, 0.0f); y1.y = fmaxf(x1.y * a + d, 0.0f);
  y1.z = fmaxf(x1.z * a + d, 0.0f); y1.w = fmaxf(x1.w * a + d, 0.0f);
  float4* yp = (float4*)(Y + (size_t)c * L);
  yp[t] = y0;
  yp[t + 256] = y1;
}

// softmax_eps over 1024-rows, single pass: 256 threads x 1 float4 in registers
__global__ void k_softmax_rows(float* __restrict__ S)
{
  const int L = 1024;
  size_t row = blockIdx.x;
  float4* p = (float4*)(S + row * (size_t)L);
  int t = threadIdx.x;                       // 256
  __shared__ float red[256];
  float4 x = p[t];
  x.x = __expf(fminf(x.x, 50.0f));
  x.y = __expf(fminf(x.y, 50.0f));
  x.z = __expf(fminf(x.z, 50.0f));
  x.w = __expf(fminf(x.w, 50.0f));
  red[t] = x.x + x.y + x.z + x.w;
  __syncthreads();
  for (int s = 128; s > 0; s >>= 1) { if (t < s) red[t] += red[t + s]; __syncthreads(); }
  float inv = 1.0f / (red[0] + 1e-12f);
  x.x *= inv; x.y *= inv; x.z *= inv; x.w *= inv;
  p[t] = x;
}

__global__ void k_split_add(const float* __restrict__ augc, const float* __restrict__ pe,
                            float* __restrict__ a1, float* __restrict__ a2)
{
  int i = blockIdx.x * blockDim.x + threadIdx.x;
  if (i >= 128 * 1024) return;
  int c = i >> 10, n = i & 1023;
  a1[i] = augc[(size_t)c * 2048 + n]        + pe[(size_t)c * 2048 + n];
  a2[i] = augc[(size_t)c * 2048 + 1024 + n] + pe[(size_t)c * 2048 + 1024 + n];
}

__global__ void k_concat_ch(const float* __restrict__ A, const float* __restrict__ Bm,
                            float* __restrict__ Z)
{
  int i = blockIdx.x * blockDim.x + threadIdx.x;
  if (i >= 256 * 1024) return;
  int c = i >> 10, n = i & 1023;
  Z[i] = (c < 128) ? A[(size_t)c * 1024 + n] : Bm[(size_t)(c - 128) * 1024 + n];
}

__global__ void k_concat_cols(const float* __restrict__ A, const float* __restrict__ Bm,
                              float* __restrict__ Z)
{
  int i = blockIdx.x * blockDim.x + threadIdx.x;
  if (i >= 128 * 2048) return;
  int c = i >> 11, n = i & 2047;
  Z[i] = (n < 1024) ? A[(size_t)c * 1024 + n] : Bm[(size_t)c * 1024 + (n - 1024)];
}

__global__ void k_fill_bins(float* __restrict__ cpl, const float* __restrict__ alpha)
{
  float a = *alpha;
  int i = blockIdx.x * blockDim.x + threadIdx.x;
  if (i < 1024) {
    cpl[(size_t)i * CPAD + 1024] = a;
    cpl[(size_t)1024 * CPAD + i] = a;
  }
  if (i == 0) cpl[(size_t)1024 * CPAD + 1024] = a;
}

__global__ void k_zero(float* __restrict__ p, int n)
{
  int i = blockIdx.x * blockDim.x + threadIdx.x;
  if (i < n) p[i] = 0.0f;
}

// tiled transpose of the 1025x1025 coupling block (stride CPAD both sides)
__global__ void k_transpose(const float* __restrict__ A, float* __restrict__ B)
{
  __shared__ float tile[32][33];
  int bx = blockIdx.x * 32, by = blockIdx.y * 32;
  int tx = threadIdx.x & 31, ty = threadIdx.x >> 5;   // 32 x 8
  for (int j = ty; j < 32; j += 8) {
    int r = by + j, c = bx + tx;
    tile[j][tx] = (r < WROW && c < WROW) ? A[(size_t)r * CPAD + c] : 0.0f;
  }
  __syncthreads();
  for (int j = ty; j < 32; j += 8) {
    int r = bx + j, c = by + tx;
    if (r < WROW && c < WROW) B[(size_t)r * CPAD + c] = tile[tx][j];
  }
}

// out[r] = lbase/llast - logsumexp_i(M[r,i] + add[i]), rows of 1025 (stride CPAD)
// 256 threads x 1 float4 (elems 0..1023) + scalar tail elem 1024 (thread 0).
__global__ void k_lse(const float* __restrict__ M, const float* __restrict__ add,
                      float* __restrict__ outv, float lbase, float llast)
{
  int r = blockIdx.x;                        // 0..1024
  int t = threadIdx.x;                       // 256
  __shared__ float red[256];
  const float* row = M + (size_t)r * CPAD;
  float4 x = ((const float4*)row)[t];
  float4 a = ((const float4*)add)[t];
  float e0 = x.x + a.x, e1 = x.y + a.y, e2 = x.z + a.z, e3 = x.w + a.w;
  float tail = row[1024] + add[1024];
  float mx = fmaxf(fmaxf(fmaxf(e0, e1), fmaxf(e2, e3)), tail);
  red[t] = mx;
  __syncthreads();
  for (int s = 128; s > 0; s >>= 1) { if (t < s) red[t] = fmaxf(red[t], red[t + s]); __syncthreads(); }
  mx = red[0];
  __syncthreads();
  float sum = __expf(e0 - mx) + __expf(e1 - mx) + __expf(e2 - mx) + __expf(e3 - mx);
  if (t == 0) sum += __expf(tail - mx);
  red[t] = sum;
  __syncthreads();
  for (int s = 128; s > 0; s >>= 1) { if (t < s) red[t] += red[t + s]; __syncthreads(); }
  if (t == 0) {
    float lm = (r == 1024) ? llast : lbase;
    outv[r] = lm - (mx + __logf(red[0]));
  }
}

__global__ void k_final(const float* __restrict__ cpl, const float* __restrict__ u,
                        const float* __restrict__ v, float* __restrict__ out, float addc)
{
  int i = blockIdx.x * blockDim.x + threadIdx.x;
  const int total = WROW * WROW;
  if (i >= total) return;
  int r = i / WROW, c = i % WROW;
  out[i] = __expf(cpl[(size_t)r * CPAD + c] + u[r] + v[c] + addc);
}

// ---------------------------------------------------------------------------
// Host orchestration
// ---------------------------------------------------------------------------
extern "C" void kernel_launch(void* const* d_in, const int* in_sizes, int n_in,
                              void* d_out, int out_size, void* d_ws, size_t ws_size,
                              hipStream_t stream)
{
  (void)in_sizes; (void)n_in; (void)out_size; (void)ws_size;
  auto P = [&](int i) -> const float* { return (const float*)d_in[i]; };

  const float* desc1 = P(0);
  const float* desc2 = P(1);
  const float* x1 = P(2);
  const float* x2 = P(3);

  // params flattened in jax-pytree order (dict keys sorted recursively):
  //   cross_blocks[9]{Wa0,Wa1,Wk,Wmh,Wq,Wv,ba0,ba1,be,bk,bmh,bq,bv,ga,lam,rbias},
  //   dc1{W0,W1,W2,b0,b1,bias2,g0,g1}, dc2{...}, dmp{W0,W1,b0,bias1,g0},
  //   dustbin, fp_W, fp_b, pmp{...}, pos_enc{W0..W4,b0..b3,bias4,g0..g3},
  //   rpp{W0..W3,b0..b2,bias3,g0..g2}, self_blocks[9]{...}
  const int CROSS0 = 4;
  const int DC1 = 148, DC2 = 156;
  const int DMP = 164;
  const int DUSTBIN = 169;
  const int FPW = 170, FPB = 171;
  const int PMP = 172;
  const int PENC = 177;
  const int RPP = 191;
  const int SELF0 = 202;
  enum { AWa0 = 0, AWa1, AWk, AWmh, AWq, AWv, Aba0, Aba1, Abe, Abk, Abmh,
         Abq, Abv, Aga, Alam, Arbias };

  // ---- workspace carve (floats) ----
  float* ws = (float*)d_ws;
  size_t off = 0;
  auto alloc = [&](size_t n) { float* p = ws + off; off += (n + 63) & ~(size_t)63; return p; };
  float* dcat   = alloc(128 * 2048);
  float* ex     = alloc(2 * 2048);
  float* tA     = alloc(256 * 2048);
  float* tB     = alloc(256 * 2048);
  float* pe     = alloc(128 * 2048);
  float* posemb = alloc(128 * 2048);
  float* augc   = alloc(128 * 2048);
  float* aug1   = alloc(128 * 1024);
  float* aug2   = alloc(128 * 1024);
  float* descp  = alloc(128 * 2048);
  float* crossR = alloc((size_t)1024 * 1024);
  float* selfR1 = alloc((size_t)1024 * 1024);
  float* selfR2 = alloc((size_t)1024 * 1024);
  float* Q1 = alloc(128 * 1024); float* K1 = alloc(128 * 1024); float* V1 = alloc(128 * 1024);
  float* Q2 = alloc(128 * 1024); float* K2 = alloc(128 * 1024); float* V2 = alloc(128 * 1024);
  float* S1  = alloc((size_t)4 * 1024 * 1024);
  float* S2  = alloc((size_t)4 * 1024 * 1024);
  float* AV1 = alloc(128 * 1024); float* AV2 = alloc(128 * 1024);
  float* M1  = alloc(128 * 1024); float* M2  = alloc(128 * 1024);
  float* cat1 = alloc(256 * 1024); float* cat2 = alloc(256 * 1024);
  float* hid  = alloc(256 * 1024);
  float* mid  = alloc(128 * 2048);
  float* dmbuf = alloc(128 * 2048);
  float* f12  = alloc(128 * 2048);
  float* cpl  = alloc((size_t)WROW * CPAD);
  float* cplT = alloc((size_t)WROW * CPAD);
  float* uvec = alloc(WROW + 15);
  float* vvec = alloc(WROW + 15);
  float* fscale = alloc(256);
  float* fshift = alloc(256);

  // ---- launch helpers ----
  auto gemm_plain = [&](const float* W, const float* X, float* Y, int Mo, int Kc, int L) {
    int tiles = (Mo / 16) * (L / 16);
    k_conv_gemm<false, false, false, false>
        <<<dim3((tiles + 7) / 8), dim3(256), 0, stream>>>(W, X, Y, nullptr, nullptr, nullptr,
                                                          Mo, Kc, L);
  };
  auto gemm_bias = [&](const float* W, const float* X, float* Y, int Mo, int Kc, int L,
                       const float* bias) {
    int tiles = (Mo / 16) * (L / 16);
    k_conv_gemm<false, true, false, false>
        <<<dim3((tiles + 7) / 8), dim3(256), 0, stream>>>(W, X, Y, nullptr, bias, nullptr,
                                                          Mo, Kc, L);
  };
  auto gemm_bnrelu = [&](const float* W, const float* X, float* Y, int Mo, int Kc, int L) {
    int tiles = (Mo / 16) * (L / 16);
    k_conv_gemm<true, true, true, false>
        <<<dim3((tiles + 7) / 8), dim3(256), 0, stream>>>(W, X, Y, fscale, fshift, nullptr,
                                                          Mo, Kc, L);
  };
  auto gemm_bias_res = [&](const float* W, const float* X, float* Y, int Mo, int Kc, int L,
                           const float* bias, const float* base) {
    int tiles = (Mo / 16) * (L / 16);
    k_conv_gemm<false, true, false, true>
        <<<dim3((tiles + 7) / 8), dim3(256), 0, stream>>>(W, X, Y, nullptr, bias, base,
                                                          Mo, Kc, L);
  };
  auto fusebn = [&](const float* g, const float* b, const float* bias, int n) {
    k_fuse_bn<<<dim3((n + 63) / 64), dim3(64), 0, stream>>>(g, b, bias, fscale, fshift, n);
  };
  auto bnconv = [&](const float* W, const float* X, float* Y, int Mo, int Kc, int L,
                    const float* g, const float* b) {
    fusebn(g, b, nullptr, Mo);
    if (Kc == 2) {
      k_conv2<<<dim3((Mo * L + 255) / 256), dim3(256), 0, stream>>>(W, X, Y, fscale, fshift, Mo, L);
    } else {
      gemm_bnrelu(W, X, Y, Mo, Kc, L);
    }
  };
  auto xty = [&](const float* X, const float* Y, float* O, int Kc, int ldx, int ldy,
                 int ldo, float sc, int meye) {
    k_xty<<<dim3(512), dim3(256), 0, stream>>>(X, Y, O, Kc, ldx, ldy, ldo, sc, meye);
  };

  // ---- preamble ----
  k_normalize_t<<<dim3(1024), dim3(128), 0, stream>>>(desc1, dcat, 128, 2048, 0);
  k_normalize_t<<<dim3(1024), dim3(128), 0, stream>>>(desc2, dcat, 128, 2048, 1024);
  k_build_ex<<<dim3(4), dim3(256), 0, stream>>>(x1, x2, ex);

  // pos encoder: 2 -> 32 -> 64 -> 128 -> 256 (bn+relu) -> 128 (+bias)
  bnconv(P(PENC + 0), ex, tA, 32, 2, 2048, P(PENC + 10), P(PENC + 5));
  bnconv(P(PENC + 1), tA, tB, 64, 32, 2048, P(PENC + 11), P(PENC + 6));
  bnconv(P(PENC + 2), tB, tA, 128, 64, 2048, P(PENC + 12), P(PENC + 7));
  bnconv(P(PENC + 3), tA, tB, 256, 128, 2048, P(PENC + 13), P(PENC + 8));
  gemm_bias(P(PENC + 4), tB, pe, 128, 256, 2048, P(PENC + 9));

  // res pos proj: 2 -> 32 -> 64 -> 128 (bn+relu) -> 128 (+bias)
  bnconv(P(RPP + 0), ex, tA, 32, 2, 2048, P(RPP + 8), P(RPP + 4));
  bnconv(P(RPP + 1), tA, tB, 64, 32, 2048, P(RPP + 9), P(RPP + 5));
  bnconv(P(RPP + 2), tB, tA, 128, 64, 2048, P(RPP + 10), P(RPP + 6));
  gemm_bias(P(RPP + 3), tA, posemb, 128, 128, 2048, P(RPP + 7));

  // desc compressors (conv -> inorm -> bn -> relu) x2 -> conv+bias
  auto compressor = [&](int base, float* out) {
    gemm_plain(P(base + 0), dcat, tA, 256, 128, 2048);
    k_inorm_bn_relu<<<dim3(256), dim3(256), 0, stream>>>(tA, tB, P(base + 6), P(base + 3));
    gemm_plain(P(base + 1), tB, tA, 256, 256, 2048);
    k_inorm_bn_relu<<<dim3(256), dim3(256), 0, stream>>>(tA, tB, P(base + 7), P(base + 4));
    gemm_bias(P(base + 2), tB, out, 128, 256, 2048, P(base + 5));
  };
  compressor(DC1, augc);
  compressor(DC2, descp);
  k_split_add<<<dim3((128 * 1024) / 256), dim3(256), 0, stream>>>(augc, pe, aug1, aug2);

  const float rs = 1.0f / sqrtf(128.0f);     // (C^-0.25)^2
  xty(descp, descp + 1024, crossR, 128, 2048, 2048, 1024, rs, 0);
  xty(posemb, posemb, selfR1, 128, 2048, 2048, 1024, rs, 1);
  xty(posemb + 1024, posemb + 1024, selfR2, 128, 2048, 2048, 1024, rs, 1);

  // mid-projection residual refresh (layer 4)
  auto midupdate = [&]() {
    k_concat_cols<<<dim3((128 * 2048) / 256), dim3(256), 0, stream>>>(aug1, aug2, mid);
    bnconv(P(DMP + 0), mid, tA, 256, 128, 2048, P(DMP + 4), P(DMP + 2));
    gemm_bias(P(DMP + 1), tA, dmbuf, 128, 256, 2048, P(DMP + 3));
    xty(dmbuf, dmbuf + 1024, crossR, 128, 2048, 2048, 1024, rs, 0);
    bnconv(P(PMP + 0), mid, tA, 256, 128, 2048, P(PMP + 4), P(PMP + 2));
    gemm_bias(P(PMP + 1), tA, dmbuf, 128, 256, 2048, P(PMP + 3));
    xty(dmbuf, dmbuf, selfR1, 128, 2048, 2048, 1024, rs, 1);
    xty(dmbuf + 1024, dmbuf + 1024, selfR2, 128, 2048, 2048, 1024, rs, 1);
  };

  // attention block
  auto attn = [&](int pb, int is_self, const float* R1, int R1t,
                  const float* R2, int R2t) {
    gemm_bias(P(pb + AWq), aug1, Q1, 128, 128, 1024, P(pb + Abq));
    gemm_bias(P(pb + AWk), aug1, K1, 128, 128, 1024, P(pb + Abk));
    gemm_bias(P(pb + AWv), aug1, V1, 128, 128, 1024, P(pb + Abv));
    gemm_bias(P(pb + AWq), aug2, Q2, 128, 128, 1024, P(pb + Abq));
    gemm_bias(P(pb + AWk), aug2, K2, 128, 128, 1024, P(pb + Abk));
    gemm_bias(P(pb + AWv), aug2, V2, 128, 128, 1024, P(pb + Abv));
    const float sc = 1.0f / sqrtf(32.0f);
    const float* lam = P(pb + Alam);
    const float* rb  = P(pb + Arbias);
    const float* Ka = is_self ? K1 : K2;
    const float* Kb = is_self ? K2 : K1;
    if (R1t) k_attn_scores<true><<<dim3(512, 1, 4), dim3(256), 0, stream>>>(Q1, Ka, R1, lam, rb, S1, sc);
    else     k_attn_scores<false><<<dim3(512, 1, 4), dim3(256), 0, stream>>>(Q1, Ka, R1, lam, rb, S1, sc);
    if (R2t) k_attn_scores<true><<<dim3(512, 1, 4), dim3(256), 0, stream>>>(Q2, Kb, R2, lam, rb, S2, sc);
    else     k_attn_scores<false><<<dim3(512, 1, 4), dim3(256), 0, stream>>>(Q2, Kb, R2, lam, rb, S2, sc);
    k_softmax_rows<<<dim3(4096), dim3(256), 0, stream>>>(S1);
    k_softmax_rows<<<dim3(4096), dim3(256), 0, stream>>>(S2);
    const float* Va = is_self ? V1 : V2;
    const float* Vb = is_self ? V2 : V1;
    k_attn_av<<<dim3(16, 1, 4), dim3(256), 0, stream>>>(S1, Va, AV1);
    k_attn_av<<<dim3(16, 1, 4), dim3(256), 0, stream>>>(S2, Vb, AV2);
    gemm_bias(P(pb + AWmh), AV1, M1, 128, 128, 1024, P(pb + Abmh));
    gemm_bias(P(pb + AWmh), AV2, M2, 128, 128, 1024, P(pb + Abmh));
    k_concat_ch<<<dim3((256 * 1024) / 256), dim3(256), 0, stream>>>(aug1, M1, cat1);
    k_concat_ch<<<dim3((256 * 1024) / 256), dim3(256), 0, stream>>>(aug2, M2, cat2);
    fusebn(P(pb + Aga), P(pb + Abe), P(pb + Aba0), 256);
    gemm_bnrelu(P(pb + AWa0), cat1, hid, 256, 256, 1024);
    gemm_bias_res(P(pb + AWa1), hid, aug1, 128, 256, 1024, P(pb + Aba1), aug1);
    gemm_bnrelu(P(pb + AWa0), cat2, hid, 256, 256, 1024);
    gemm_bias_res(P(pb + AWa1), hid, aug2, 128, 256, 1024, P(pb + Aba1), aug2);
  };

  for (int i = 0; i < 9; ++i) {
    if (i == 4) midupdate();
    attn(SELF0 + i * 16, 1, selfR1, 0, selfR2, 0);
    attn(CROSS0 + i * 16, 0, crossR, 0, crossR, 1);
  }

  // final projection + scores into coupling interior (row stride CPAD)
  gemm_bias(P(FPW), aug1, f12,              128, 128, 1024, P(FPB));
  gemm_bias(P(FPW), aug2, f12 + 128 * 1024, 128, 128, 1024, P(FPB));
  xty(f12, f12 + 128 * 1024, cpl, 128, 1024, 1024, CPAD, 1.0f / sqrtf(128.0f), 0);
  k_fill_bins<<<dim3(5), dim3(256), 0, stream>>>(cpl, P(DUSTBIN));
  k_transpose<<<dim3(33, 33), dim3(256), 0, stream>>>(cpl, cplT);

  // Sinkhorn (100 iterations); couplings constant -> both passes row-major
  k_zero<<<dim3(5), dim3(256), 0, stream>>>(uvec, WROW);
  k_zero<<<dim3(5), dim3(256), 0, stream>>>(vvec, WROW);
  const float lnorm = -logf(2048.0f);
  const float lmu_last = logf(1024.0f) + lnorm;
  for (int it = 0; it < 100; ++it) {
    k_lse<<<dim3(WROW), dim3(256), 0, stream>>>(cpl,  vvec, uvec, lnorm, lmu_last);
    k_lse<<<dim3(WROW), dim3(256), 0, stream>>>(cplT, uvec, vvec, lnorm, lmu_last);
  }
  k_final<<<dim3((WROW * WROW + 255) / 256), dim3(256), 0, stream>>>(
      cpl, uvec, vvec, (float*)d_out, -lnorm);
}
